// TanhBasisActivationLayerAffine_56495999811652
// MI455X (gfx1250) — matrix operand order repaired and checked by reading, not compiled, not run on hardware
//
#include <hip/hip_runtime.h>
#include <hip/hip_bf16.h>
#include <stdint.h>

typedef _Float16 h4   __attribute__((ext_vector_type(4)));
typedef _Float16 h8   __attribute__((ext_vector_type(8)));
typedef _Float16 v16h __attribute__((ext_vector_type(16)));
typedef float    v8f  __attribute__((ext_vector_type(8)));
typedef float    f4   __attribute__((ext_vector_type(4)));
typedef unsigned int u32x4 __attribute__((ext_vector_type(4)));
typedef int          i32x4 __attribute__((ext_vector_type(4)));
typedef int          i32x8 __attribute__((ext_vector_type(8)));

#define IN_F   512
#define OUT_F  512
#define KBAS   8
#define KTOT   (IN_F * KBAS)   // 4096 = GEMM K
#define BATCHN 8192
#define BM     128             // batch tile
#define BN     128             // out tile
#define KC     128             // K-chunk = 16 features * 8 basis
#define LDA    136             // padded LDS row (halves)
#define LDB    136             // = 128 + 8 halves (16B = 4 DWORD pad per row)

// ---- one-time coeffs f32 -> f16 conversion into workspace ----
__global__ __launch_bounds__(256)
void convert_coeffs_f16(const float* __restrict__ src,
                        _Float16* __restrict__ dst, int n)
{
    int i = (blockIdx.x * 256 + threadIdx.x) * 4;
    if (i < n) {
        f4 v = *(const f4*)(src + i);
        h4 hv;
#pragma unroll
        for (int e = 0; e < 4; ++e) hv[e] = (_Float16)v[e];
        *(h4*)(dst + i) = hv;
    }
}

// D = A(basis, BMxK f16, built on the fly) * B(coeffs^T, KxBN f16) + C(f32)
template <bool USE_TDM>
__global__ __launch_bounds__(256, 2)
void tanh_basis_gemm_kernel(const float* __restrict__ x,
                            const float* __restrict__ coeffs,
                            const _Float16* __restrict__ coeffsH,
                            const float* __restrict__ raw_alpha,
                            const float* __restrict__ beta,
                            const float* __restrict__ centers,
                            const float* __restrict__ slopes,
                            float* __restrict__ out)
{
    __shared__ _Float16 Al[BM * LDA];     // A tile, row-major (b, c=k-dim)
    __shared__ _Float16 Bl[BN * LDB];     // B tile, K-major per out column
    __shared__ float alphaS[IN_F];
    __shared__ float betaS[IN_F];
    __shared__ float slope2S[KTOT];       // 2*slope
    __shared__ float cen2S[KTOT];         // 2*slope*center

    const int t   = threadIdx.x;          // 0..255
    const int bm0 = blockIdx.x * BM;      // batch base
    const int on0 = blockIdx.y * BN;      // out base

    // ---- per-block precompute of per-feature params into LDS ----
    for (int i = t; i < IN_F; i += 256) {
        float ra = raw_alpha[i];
        alphaS[i] = __logf(1.0f + __expf(ra)) + 1e-6f;   // softplus
        betaS[i]  = beta[i];
    }
    for (int i = t; i < KTOT; i += 256) {
        float s = slopes[i];
        slope2S[i] = 2.0f * s;
        cen2S[i]   = 2.0f * s * centers[i];
    }
    __syncthreads();

    // wave / lane decomposition
    const int wid  = t >> 5;
    const int lane = t & 31;
    const int lm   = lane & 15;
    const int hi   = lane >> 4;
    const int wm   = (wid >> 2) * 64;     // wave row offset in tile (0,64)
    const int wn   = (wid & 3) * 32;      // wave col offset in tile (0..96)

    // Wave-uniform SGPR leader flag: guarantees a scalar branch around the
    // TDM issue (TENSOR_LOAD_TO_LDS ignores EXEC, so an exec-mask "branch"
    // would make every wave issue the DMA).
    const int wid_s = __builtin_amdgcn_readfirstlane(wid);

    v8f acc[4][2];
#pragma unroll
    for (int mi = 0; mi < 4; ++mi)
#pragma unroll
        for (int ni = 0; ni < 2; ++ni)
            acc[mi][ni] = (v8f)0.0f;

    const int arow = t >> 1;              // 0..127: batch row for A staging
    const int ah   = t & 1;               // which 8 features
    const int bcol = t >> 1;              // 0..127: out column for B staging
    const int bh   = t & 1;               // which 64 K values

#pragma unroll 1
    for (int ch = 0; ch < KTOT / KC; ++ch) {
        const int i0 = ch * (KC / KBAS);  // feature base of this chunk (16/step)

        // ---- stage B via Tensor Data Mover: 128x128 f16 2D tile -> LDS ----
        if (USE_TDM) {
            if (wid_s == 0) {
                const _Float16* gp = coeffsH + (size_t)on0 * KTOT
                                     + (size_t)i0 * KBAS;
                uint64_t ga  = (uint64_t)(uintptr_t)gp;
                uint32_t lds = (uint32_t)(uintptr_t)&Bl[0];
                u32x4 g0;
                g0[0] = 1u;                                  // count=1, user D#
                g0[1] = lds;                                 // lds_addr
                g0[2] = (uint32_t)ga;                        // global_addr[31:0]
                g0[3] = (uint32_t)(ga >> 32) | (2u << 30);   // addr[56:32]|type=2
                i32x8 g1;
                g1[0] = (1 << 16)       // data_size = 2 bytes
                      | (1 << 20)       // pad_enable (LDS row padding)
                      | (5 << 22)       // pad_interval: 64 DWORDs (=256B row)
                      | (3 << 25);      // pad_amount: 4 DWORDs (=16B)
                g1[1] = (KTOT & 0xFFFF) << 16;          // tensor_dim0[15:0]
                g1[2] = ((KTOT >> 16) & 0xFFFF)         // tensor_dim0[31:16]
                      | ((OUT_F & 0xFFFF) << 16);       // tensor_dim1[15:0]
                g1[3] = ((OUT_F >> 16) & 0xFFFF)        // tensor_dim1[31:16]
                      | (128 << 16);                    // tile_dim0 = 128
                g1[4] = 128;                            // tile_dim1=128, dim2=0
                g1[5] = KTOT;                           // dim0_stride[31:0]
                g1[6] = 0;                              // stride hi / dim1_stride
                g1[7] = 0;
                i32x4 z4 = (i32x4)0;
                i32x8 z8 = (i32x8)0;
                // clang-23 / therock signature: (g0, g1, g2, g3, ext, cpol)
                __builtin_amdgcn_tensor_load_to_lds(g0, g1, z4, z4, z8, 0);
            }
        }

        // ---- stage A: basis tile (128 x 128 f16) ----
        {
            const float* xp = x + (size_t)(bm0 + arow) * IN_F + i0 + ah * 8;
            f4 x0 = *(const f4*)(xp);
            f4 x1 = *(const f4*)(xp + 4);
            float xv[8];
#pragma unroll
            for (int j = 0; j < 4; ++j) { xv[j] = x0[j]; xv[j + 4] = x1[j]; }
#pragma unroll
            for (int j = 0; j < 8; ++j) {
                const int ig = i0 + ah * 8 + j;                 // global feature
                const float u = alphaS[ig] * xv[j] + betaS[ig]; // affine
                h8 hv;
#pragma unroll
                for (int k = 0; k < 8; ++k) {
                    // 0.5*(1+tanh(s*(u-c))) == sigmoid(2s*u - 2s*c)
                    float z = slope2S[ig * 8 + k] * u - cen2S[ig * 8 + k];
                    // sigmoid via v_exp_f32 + v_rcp_f32 (no IEEE divide)
                    float b = __builtin_amdgcn_rcpf(1.0f + __expf(-z));
                    hv[k] = (_Float16)b;
                }
                *(h8*)&Al[arow * LDA + (ah * 8 + j) * 8] = hv;
            }
        }

        // ---- stage B fallback: manual load+convert (no workspace) ----
        if (!USE_TDM) {
            const float* cp = coeffs + (size_t)(on0 + bcol) * KTOT
                              + (size_t)i0 * KBAS + bh * 64;
            if (ch + 1 < KTOT / KC)
                __builtin_prefetch(cp + KC, 0, 3);
#pragma unroll
            for (int v = 0; v < 8; ++v) {
                f4 c0 = *(const f4*)(cp + v * 8);
                f4 c1 = *(const f4*)(cp + v * 8 + 4);
                h8 hv;
#pragma unroll
                for (int e = 0; e < 4; ++e) {
                    hv[e]     = (_Float16)c0[e];
                    hv[e + 4] = (_Float16)c1[e];
                }
                *(h8*)&Bl[bcol * LDB + bh * 64 + v * 8] = hv;
            }
        }

        if (USE_TDM) {
            if (wid_s == 0)
                __builtin_amdgcn_s_wait_tensorcnt(0);  // B tile landed in LDS
        }
        __syncthreads();

        // ---- compute: 4 k-steps of 32, 8 WMMAs each ----
#pragma unroll
        for (int kk = 0; kk < KC; kk += 32) {
            v16h af[4];
            v16h bf[2];
#pragma unroll
            for (int mi = 0; mi < 4; ++mi) {
                // A 16-bit layout: lane holds K in {hi*8..+7} and {hi*8+16..+23}
                const _Float16* ap = &Al[(wm + mi * 16 + lm) * LDA + kk + hi * 8];
                union { v16h v; h8 h[2]; } u;
                u.h[0] = *(const h8*)ap;
                u.h[1] = *(const h8*)(ap + 16);
                af[mi] = u.v;
            }
#pragma unroll
            for (int ni = 0; ni < 2; ++ni) {
                // B layout: lane holds K in {hi*8..+7} and {hi*8+16..+23} (matches A)
                const _Float16* bp = &Bl[(wn + ni * 16 + lm) * LDB + kk + hi * 8];
                union { v16h v; h8 h[2]; } u;
                u.h[0] = *(const h8*)bp;
                u.h[1] = *(const h8*)(bp + 16);
                bf[ni] = u.v;
            }
#pragma unroll
            for (int mi = 0; mi < 4; ++mi)
#pragma unroll
                for (int ni = 0; ni < 2; ++ni)
                    acc[mi][ni] = __builtin_amdgcn_wmma_f32_16x16x32_f16(
                        false, af[mi], false, bf[ni],
                        (short)0, acc[mi][ni], false, false);
        }
        __syncthreads();
    }

    // ---- store: C/D layout VGPR r -> M = r + 8*hi, N = lane&15 ----
#pragma unroll
    for (int mi = 0; mi < 4; ++mi)
#pragma unroll
        for (int ni = 0; ni < 2; ++ni) {
            const int n = on0 + wn + ni * 16 + lm;
#pragma unroll
            for (int r = 0; r < 8; ++r) {
                const int m = bm0 + wm + mi * 16 + hi * 8 + r;
                out[(size_t)m * OUT_F + n] = acc[mi][ni][r];
            }
        }
}

extern "C" void kernel_launch(void* const* d_in, const int* in_sizes, int n_in,
                              void* d_out, int out_size, void* d_ws, size_t ws_size,
                              hipStream_t stream) {
    (void)in_sizes; (void)n_in; (void)out_size;
    const float* x         = (const float*)d_in[0];
    const float* coeffs    = (const float*)d_in[1];
    const float* raw_alpha = (const float*)d_in[2];
    const float* beta      = (const float*)d_in[3];
    const float* centers   = (const float*)d_in[4];
    const float* slopes    = (const float*)d_in[5];
    float* out = (float*)d_out;

    dim3 grid(BATCHN / BM, OUT_F / BN);   // (64, 4)
    const int ncoef = OUT_F * KTOT;       // 2,097,152 elements
    const size_t need = (size_t)ncoef * sizeof(_Float16);

    if (ws_size >= need) {
        _Float16* coeffsH = (_Float16*)d_ws;
        convert_coeffs_f16<<<ncoef / (256 * 4), 256, 0, stream>>>(
            coeffs, coeffsH, ncoef);
        tanh_basis_gemm_kernel<true><<<grid, 256, 0, stream>>>(
            x, coeffs, coeffsH, raw_alpha, beta, centers, slopes, out);
    } else {
        tanh_basis_gemm_kernel<false><<<grid, 256, 0, stream>>>(
            x, coeffs, nullptr, raw_alpha, beta, centers, slopes, out);
    }
}
